// SelfAttention_51565377356400
// MI455X (gfx1250) — compile-verified, matrix operand order checked by reading, not run on hardware
//
#include <hip/hip_runtime.h>
#include <hip/hip_bf16.h>

// ---------------------------------------------------------------------------
// Self-attention variant (use_xv) for MI455X / gfx1250, f16 WMMA pipeline.
//   1) cvt:   fp32 -> f16 for q,k,v activations and the 6 weight matrices
//   2) proj:  Q,K (head-major [bh][s][64]) and V,xV (transposed [bh][64][s])
//   3) flash: online-softmax attention, 32 query rows per wave; one P tile
//             feeds both P@V (v_bar) and P@xV (k_bar)
//   4) out:   x = kbar @ Wxo^T + bxo ; m = vbar @ Wmo^T + bmo  (fp32 out)
// All GEMMs are NT-shaped so every A/B fragment load is a contiguous 16-byte
// chunk matching the CDNA5 16-bit WMMA VGPR layouts.
// ---------------------------------------------------------------------------

typedef __attribute__((ext_vector_type(16))) _Float16 v16h;
typedef __attribute__((ext_vector_type(8)))  _Float16 v8h;
typedef __attribute__((ext_vector_type(4)))  _Float16 v4h;
typedef __attribute__((ext_vector_type(8)))  float    v8f;
typedef __attribute__((ext_vector_type(4)))  float    v4f;

#define D_X   1024
#define D_HD  64
#define N_I   16
#define BSZ   2
#define SEQ   2048
#define M_TOT (BSZ * SEQ)   /* 4096 */
#define NH    (BSZ * N_I)   /* 32   */

static __device__ __forceinline__ v8f vzero8() {
  v8f z = {0.f, 0.f, 0.f, 0.f, 0.f, 0.f, 0.f, 0.f};
  return z;
}

static __device__ __forceinline__ v8f wmma16(v16h a, v16h b, v8f c) {
  // D = A(16x32 f16) * B(32x16 f16) + C(16x16 f32)
  return __builtin_amdgcn_wmma_f32_16x16x32_f16(
      false, a, false, b, (short)0, c, false, false);
}

// Load a 16x32 f16 fragment (A: lane = row M; B: lane = col N) from a
// row-major matrix with leading dimension `ld` (in elements).
// Lane layout (ISA 7.12.2, 16-bit A 16x32): lanes 0-15 hold K {0..7,16..23},
// lanes 16-31 hold K {8..15,24..31}; each half is a contiguous 16-byte chunk.
static __device__ __forceinline__ v16h load_frag(const _Float16* __restrict__ base,
                                                 int rc0, int k0, int ld) {
  const int lane = (int)(threadIdx.x & 31u);
  const int hsel = lane >> 4;
  const int r    = lane & 15;
  const _Float16* p = base + (size_t)(rc0 + r) * (size_t)ld + (size_t)(k0 + hsel * 8);
  v8h lo = *(const v8h*)(p);
  v8h hi = *(const v8h*)(p + 16);
  return __builtin_shufflevector(lo, hi, 0, 1, 2, 3, 4, 5, 6, 7,
                                 8, 9, 10, 11, 12, 13, 14, 15);
}

// A-fragment load where the M x 1024 activation is stored head-major:
// element (m, k) lives at [((b*16 + k/64)*2048 + s)*64 + (k%64)],  m = b*2048+s.
static __device__ __forceinline__ v16h load_frag_hm(const _Float16* __restrict__ base,
                                                    int m0, int k0) {
  const int lane = (int)(threadIdx.x & 31u);
  const int hsel = lane >> 4;
  const int r    = lane & 15;
  const int m    = m0 + r;
  const int b    = m >> 11;
  const int s    = m & 2047;
  const int k    = k0 + hsel * 8;     // k0 is a multiple of 32 -> chunk stays in head
  const int h    = k >> 6;
  const int d0   = k & 63;
  const _Float16* p = base + ((size_t)(b * N_I + h) * SEQ + (size_t)s) * D_HD + d0;
  v8h lo = *(const v8h*)(p);
  v8h hi = *(const v8h*)(p + 16);
  return __builtin_shufflevector(lo, hi, 0, 1, 2, 3, 4, 5, 6, 7,
                                 8, 9, 10, 11, 12, 13, 14, 15);
}

// ---------------------------------------------------------------------------
// Kernel 1: fp32 -> f16 elementwise convert (vectorized x4)
// ---------------------------------------------------------------------------
__global__ void cvt_f32_f16(const float* __restrict__ src,
                            _Float16* __restrict__ dst, int n4) {
  int i = (int)(blockIdx.x * blockDim.x + threadIdx.x);
  if (i >= n4) return;
  v4f f = ((const v4f*)src)[i];
  v4h h;
#pragma unroll
  for (int r = 0; r < 4; r++) h[r] = (_Float16)f[r];
  ((v4h*)dst)[i] = h;
}

// ---------------------------------------------------------------------------
// Kernel 2: fused input projections.  z = 0:Q  1:K  2:V  3:xV
// One wave computes a 64x64 tile of Y = X @ W^T + b.
// Q,K written head-major [bh][s][64]; V,xV written transposed [bh][64][s].
// ---------------------------------------------------------------------------
__global__ __launch_bounds__(32)
void proj_gemm(const _Float16* __restrict__ q16, const _Float16* __restrict__ k16,
               const _Float16* __restrict__ v16, const _Float16* __restrict__ w16,
               const float* __restrict__ bq, const float* __restrict__ bk,
               const float* __restrict__ bv, const float* __restrict__ bxv,
               _Float16* __restrict__ Qhm, _Float16* __restrict__ Khm,
               _Float16* __restrict__ VT,  _Float16* __restrict__ xVT) {
  const int z = (int)blockIdx.z;
  const _Float16* A;
  const _Float16* W;
  const float* bias;
  _Float16* out;
  bool transposed;
  const size_t SZ_W = (size_t)D_X * D_X;
  switch (z) {
    case 0:  A = q16; W = w16 + 0 * SZ_W; bias = bq;  out = Qhm; transposed = false; break;
    case 1:  A = k16; W = w16 + 1 * SZ_W; bias = bk;  out = Khm; transposed = false; break;
    case 2:  A = v16; W = w16 + 2 * SZ_W; bias = bv;  out = VT;  transposed = true;  break;
    default: A = k16; W = w16 + 3 * SZ_W; bias = bxv; out = xVT; transposed = true;  break;
  }
  const int m0 = (int)blockIdx.x * 64;
  const int n0 = (int)blockIdx.y * 64;

  v8f acc[4][4];
#pragma unroll
  for (int i = 0; i < 4; i++)
#pragma unroll
    for (int j = 0; j < 4; j++) acc[i][j] = vzero8();

  for (int k0 = 0; k0 < D_X; k0 += 32) {
    v16h a[4], b[4];
#pragma unroll
    for (int i = 0; i < 4; i++) a[i] = load_frag(A, m0 + 16 * i, k0, D_X);
#pragma unroll
    for (int j = 0; j < 4; j++) b[j] = load_frag(W, n0 + 16 * j, k0, D_X);
#pragma unroll
    for (int i = 0; i < 4; i++)
#pragma unroll
      for (int j = 0; j < 4; j++) acc[i][j] = wmma16(a[i], b[j], acc[i][j]);
  }

  const int lane = (int)(threadIdx.x & 31u);
  const int hsel = lane >> 4;
  const int ln   = lane & 15;
  const int b_   = m0 >> 11;  // batch (tile never crosses batch: 2048 % 64 == 0)

#pragma unroll
  for (int i = 0; i < 4; i++) {
    const int s_base = ((m0 + 16 * i) & 2047) + hsel * 8;
#pragma unroll
    for (int j = 0; j < 4; j++) {
      const int col = n0 + 16 * j + ln;          // output feature = h*64 + d
      const float bj = bias[col];
      const int h = col >> 6;
      const int d = col & 63;
      if (!transposed) {
        _Float16* o = out + ((size_t)(b_ * N_I + h) * SEQ) * D_HD + d;
#pragma unroll
        for (int r = 0; r < 8; r++)
          o[(size_t)(s_base + r) * D_HD] = (_Float16)(acc[i][j][r] + bj);
      } else {
        v8h pk;
#pragma unroll
        for (int r = 0; r < 8; r++) pk[r] = (_Float16)(acc[i][j][r] + bj);
        *(v8h*)(out + ((size_t)(b_ * N_I + h) * D_HD + d) * SEQ + s_base) = pk;
      }
    }
  }
}

// ---------------------------------------------------------------------------
// Kernel 3: flash attention.  One wave = 32 query rows (2 row-chunks of 16)
// of one (b,h).  Per 32-column j-block: 8 WMMAs for S = Q K^T (K fragments
// shared across row-chunks), online softmax, P -> LDS transpose, then 16
// WMMAs for P@V (-> v_bar) and P@xV (-> k_bar) with V/xV fragments shared
// across row-chunks.  24 WMMAs per 24 16-byte global loads per iteration.
// ---------------------------------------------------------------------------
__global__ __launch_bounds__(32)
void flash_attn(const _Float16* __restrict__ Qhm, const _Float16* __restrict__ Khm,
                const _Float16* __restrict__ VT,  const _Float16* __restrict__ xVT,
                _Float16* __restrict__ vbar, _Float16* __restrict__ kbar) {
  __shared__ _Float16 Pl[2][16 * 32];

  const int bh = (int)blockIdx.y;
  const int s0 = (int)blockIdx.x * 32;
  const _Float16* Q  = Qhm + (size_t)bh * SEQ * D_HD;
  const _Float16* K  = Khm + (size_t)bh * SEQ * D_HD;
  const _Float16* Vt = VT  + (size_t)bh * D_HD * SEQ;
  const _Float16* Xt = xVT + (size_t)bh * D_HD * SEQ;

  v16h qa[2][2];
#pragma unroll
  for (int rc = 0; rc < 2; rc++) {
    qa[rc][0] = load_frag(Q, s0 + 16 * rc, 0, D_HD);
    qa[rc][1] = load_frag(Q, s0 + 16 * rc, 32, D_HD);
  }

  v8f ov[2][4], ox[2][4];
#pragma unroll
  for (int rc = 0; rc < 2; rc++)
#pragma unroll
    for (int t = 0; t < 4; t++) { ov[rc][t] = vzero8(); ox[rc][t] = vzero8(); }
  float mrow[2][8], lrow[2][8];
#pragma unroll
  for (int rc = 0; rc < 2; rc++)
#pragma unroll
    for (int r = 0; r < 8; r++) { mrow[rc][r] = -1.0e30f; lrow[rc][r] = 0.f; }

  const int lane = (int)(threadIdx.x & 31u);
  const int hsel = lane >> 4;
  const int ln   = lane & 15;

  for (int j0 = 0; j0 < SEQ; j0 += 32) {
    // ---- K fragments for this j-block (shared by both row-chunks)
    v16h kb[2][2];
    kb[0][0] = load_frag(K, j0, 0, D_HD);
    kb[0][1] = load_frag(K, j0, 32, D_HD);
    kb[1][0] = load_frag(K, j0 + 16, 0, D_HD);
    kb[1][1] = load_frag(K, j0 + 16, 32, D_HD);

    // ---- S tiles (2 row-chunks x 32 cols), k = d = 64 in two WMMA k-steps
    v8f sc[2][2];
#pragma unroll
    for (int rc = 0; rc < 2; rc++)
#pragma unroll
      for (int cc = 0; cc < 2; cc++) {
        v8f s = vzero8();
        s = wmma16(qa[rc][0], kb[cc][0], s);
        s = wmma16(qa[rc][1], kb[cc][1], s);
        sc[rc][cc] = s;
      }

    // ---- online softmax per row-chunk; P -> LDS
#pragma unroll
    for (int rc = 0; rc < 2; rc++) {
#pragma unroll
      for (int r = 0; r < 8; r++) { sc[rc][0][r] *= 0.125f; sc[rc][1][r] *= 0.125f; }
      // row max over 32 cols (butterfly within each 16-lane half)
      float t[8];
#pragma unroll
      for (int r = 0; r < 8; r++) t[r] = fmaxf(sc[rc][0][r], sc[rc][1][r]);
#pragma unroll
      for (int msk = 1; msk < 16; msk <<= 1)
#pragma unroll
        for (int r = 0; r < 8; r++) t[r] = fmaxf(t[r], __shfl_xor(t[r], msk, 32));

      float scl[8];
#pragma unroll
      for (int r = 0; r < 8; r++) {
        float mn = fmaxf(mrow[rc][r], t[r]);
        scl[r] = __expf(mrow[rc][r] - mn);
        mrow[rc][r] = mn;
      }
#pragma unroll
      for (int r = 0; r < 8; r++) {
        sc[rc][0][r] = __expf(sc[rc][0][r] - mrow[rc][r]);
        sc[rc][1][r] = __expf(sc[rc][1][r] - mrow[rc][r]);
      }
      // row sum
      float u[8];
#pragma unroll
      for (int r = 0; r < 8; r++) u[r] = sc[rc][0][r] + sc[rc][1][r];
#pragma unroll
      for (int msk = 1; msk < 16; msk <<= 1)
#pragma unroll
        for (int r = 0; r < 8; r++) u[r] += __shfl_xor(u[r], msk, 32);
#pragma unroll
      for (int r = 0; r < 8; r++) lrow[rc][r] = lrow[rc][r] * scl[r] + u[r];
      // rescale running accumulators
#pragma unroll
      for (int tt = 0; tt < 4; tt++)
#pragma unroll
        for (int r = 0; r < 8; r++) { ov[rc][tt][r] *= scl[r]; ox[rc][tt][r] *= scl[r]; }

      // P (C-layout) -> LDS row-major [16][32] f16
      const int row0 = hsel * 8;
#pragma unroll
      for (int r = 0; r < 8; r++) {
        Pl[rc][(row0 + r) * 32 + ln]      = (_Float16)sc[rc][0][r];
        Pl[rc][(row0 + r) * 32 + 16 + ln] = (_Float16)sc[rc][1][r];
      }
    }
    asm volatile("s_wait_dscnt 0" ::: "memory");

    // ---- read P as 16x32 A fragments from LDS
    v16h pa[2];
#pragma unroll
    for (int rc = 0; rc < 2; rc++) {
      const _Float16* pp = &Pl[rc][ln * 32 + hsel * 8];
      v8h lo = *(const v8h*)(pp);
      v8h hi = *(const v8h*)(pp + 16);
      pa[rc] = __builtin_shufflevector(lo, hi, 0, 1, 2, 3, 4, 5, 6, 7,
                                       8, 9, 10, 11, 12, 13, 14, 15);
    }

    // ---- O_v += P @ V, O_xv += P @ xV (V/xV fragments shared across rcs)
#pragma unroll
    for (int tt = 0; tt < 4; tt++) {
      v16h vb = load_frag(Vt, tt * 16, j0, SEQ);
      v16h xb = load_frag(Xt, tt * 16, j0, SEQ);
#pragma unroll
      for (int rc = 0; rc < 2; rc++) {
        ov[rc][tt] = wmma16(pa[rc], vb, ov[rc][tt]);
        ox[rc][tt] = wmma16(pa[rc], xb, ox[rc][tt]);
      }
    }
  }

  // ---- normalize and store head-major f16: vbar <- O_v, kbar <- O_xv
  _Float16* vb = vbar + (size_t)bh * SEQ * D_HD;
  _Float16* kb = kbar + (size_t)bh * SEQ * D_HD;
#pragma unroll
  for (int rc = 0; rc < 2; rc++) {
    float inv[8];
#pragma unroll
    for (int r = 0; r < 8; r++) inv[r] = 1.0f / lrow[rc][r];
#pragma unroll
    for (int tt = 0; tt < 4; tt++) {
      const int d = tt * 16 + ln;
#pragma unroll
      for (int r = 0; r < 8; r++) {
        const int s = s0 + 16 * rc + hsel * 8 + r;
        vb[(size_t)s * D_HD + d] = (_Float16)(ov[rc][tt][r] * inv[r]);
        kb[(size_t)s * D_HD + d] = (_Float16)(ox[rc][tt][r] * inv[r]);
      }
    }
  }
}

// ---------------------------------------------------------------------------
// Kernel 4: fused output projections.  z = 0: x = kbar@Wxo^T+bxo
//                                      z = 1: m = vbar@Wmo^T+bmo
// A is head-major f16; output fp32 straight into d_out.
// ---------------------------------------------------------------------------
__global__ __launch_bounds__(32)
void out_gemm(const _Float16* __restrict__ kbar, const _Float16* __restrict__ vbar,
              const _Float16* __restrict__ w16, const float* __restrict__ bxo,
              const float* __restrict__ bmo, float* __restrict__ out) {
  const int z = (int)blockIdx.z;
  const size_t SZ_W = (size_t)D_X * D_X;
  const _Float16* A = z ? vbar : kbar;
  const _Float16* W = w16 + (size_t)(4 + z) * SZ_W;
  const float* bias = z ? bmo : bxo;
  float* O = out + (size_t)z * ((size_t)M_TOT * D_X);

  const int m0 = (int)blockIdx.x * 64;
  const int n0 = (int)blockIdx.y * 64;

  v8f acc[4][4];
#pragma unroll
  for (int i = 0; i < 4; i++)
#pragma unroll
    for (int j = 0; j < 4; j++) acc[i][j] = vzero8();

  for (int k0 = 0; k0 < D_X; k0 += 32) {
    v16h a[4], b[4];
#pragma unroll
    for (int i = 0; i < 4; i++) a[i] = load_frag_hm(A, m0 + 16 * i, k0);
#pragma unroll
    for (int j = 0; j < 4; j++) b[j] = load_frag(W, n0 + 16 * j, k0, D_X);
#pragma unroll
    for (int i = 0; i < 4; i++)
#pragma unroll
      for (int j = 0; j < 4; j++) acc[i][j] = wmma16(a[i], b[j], acc[i][j]);
  }

  const int lane = (int)(threadIdx.x & 31u);
  const int hsel = lane >> 4;
  const int ln   = lane & 15;
#pragma unroll
  for (int i = 0; i < 4; i++) {
    const int mb = m0 + 16 * i + hsel * 8;
#pragma unroll
    for (int j = 0; j < 4; j++) {
      const int col = n0 + 16 * j + ln;
      const float bj = bias[col];
#pragma unroll
      for (int r = 0; r < 8; r++)
        O[(size_t)(mb + r) * D_X + col] = acc[i][j][r] + bj;
    }
  }
}

// ---------------------------------------------------------------------------
extern "C" void kernel_launch(void* const* d_in, const int* in_sizes, int n_in,
                              void* d_out, int out_size, void* d_ws, size_t ws_size,
                              hipStream_t stream) {
  const float* q   = (const float*)d_in[0];
  const float* k   = (const float*)d_in[1];
  const float* v   = (const float*)d_in[2];
  const float* Wq  = (const float*)d_in[3];
  const float* bq  = (const float*)d_in[4];
  const float* Wk  = (const float*)d_in[5];
  const float* bk  = (const float*)d_in[6];
  const float* Wv  = (const float*)d_in[7];
  const float* bv  = (const float*)d_in[8];
  const float* Wxv = (const float*)d_in[9];
  const float* bxv = (const float*)d_in[10];
  const float* Wxo = (const float*)d_in[11];
  const float* bxo = (const float*)d_in[12];
  const float* Wmo = (const float*)d_in[13];
  const float* bmo = (const float*)d_in[14];
  float* out = (float*)d_out;

  const size_t SZ_ACT = (size_t)M_TOT * D_X;  // 4,194,304 elements
  const size_t SZ_W   = (size_t)D_X * D_X;    // 1,048,576 elements
  _Float16* ws  = (_Float16*)d_ws;
  _Float16* q16 = ws;
  _Float16* k16 = q16 + SZ_ACT;
  _Float16* v16 = k16 + SZ_ACT;
  _Float16* w16 = v16 + SZ_ACT;               // 6 weight matrices
  _Float16* Qhm = w16 + 6 * SZ_W;
  _Float16* Khm = Qhm + SZ_ACT;
  _Float16* VT  = Khm + SZ_ACT;
  _Float16* xVT = VT + SZ_ACT;
  _Float16* kbar = xVT + SZ_ACT;
  _Float16* vbar = kbar + SZ_ACT;

  // 1) fp32 -> f16 conversions
  {
    struct Job { const float* s; _Float16* d; size_t n; } jobs[9] = {
        {q, q16, SZ_ACT},   {k, k16, SZ_ACT},   {v, v16, SZ_ACT},
        {Wq, w16 + 0 * SZ_W, SZ_W}, {Wk, w16 + 1 * SZ_W, SZ_W},
        {Wv, w16 + 2 * SZ_W, SZ_W}, {Wxv, w16 + 3 * SZ_W, SZ_W},
        {Wxo, w16 + 4 * SZ_W, SZ_W}, {Wmo, w16 + 5 * SZ_W, SZ_W}};
    for (int i = 0; i < 9; i++) {
      int n4 = (int)(jobs[i].n / 4);
      cvt_f32_f16<<<(n4 + 255) / 256, 256, 0, stream>>>(jobs[i].s, jobs[i].d, n4);
    }
  }

  // 2) input projections (Q, K normal head-major; V, xV transposed)
  proj_gemm<<<dim3(M_TOT / 64, D_X / 64, 4), 32, 0, stream>>>(
      q16, k16, v16, w16, bq, bk, bv, bxv, Qhm, Khm, VT, xVT);

  // 3) flash attention -> vbar (attention@V), kbar (attention@xV)
  flash_attn<<<dim3(SEQ / 32, NH), 32, 0, stream>>>(Qhm, Khm, VT, xVT, vbar, kbar);

  // 4) output projections -> d_out (x at offset 0, m after it)
  out_gemm<<<dim3(M_TOT / 64, D_X / 64, 2), 32, 0, stream>>>(
      kbar, vbar, w16, bxo, bmo, out);
}
